// SlidingWindowSelfAttention_66451734004326
// MI455X (gfx1250) — matrix operand order repaired and checked by reading, not compiled, not run on hardware
//
#include <hip/hip_runtime.h>
#include <hip/hip_bf16.h>

typedef __attribute__((ext_vector_type(16))) _Float16 v16h;
typedef __attribute__((ext_vector_type(8)))  _Float16 v8h;
typedef __attribute__((ext_vector_type(8)))  float    v8f;
typedef __attribute__((ext_vector_type(4)))  unsigned v4u;
typedef __attribute__((ext_vector_type(8)))  int      v8i;
typedef __attribute__((ext_vector_type(4)))  int      v4i;

#define S_LEN 4096
#define HEADS 12
#define DHEAD 64
#define EMB   768
#define WIN   128
#define BATCH 2

#if defined(__has_builtin)
#if __has_builtin(__builtin_amdgcn_tensor_load_to_lds) && __has_builtin(__builtin_amdgcn_s_wait_tensorcnt)
#define USE_TDM 1
#endif
#endif
#ifndef USE_TDM
#define USE_TDM 0
#endif

static __device__ __forceinline__ v8f wmma_f16(v16h a, v16h b, v8f c) {
  return __builtin_amdgcn_wmma_f32_16x16x32_f16(
      /*neg_a=*/false, a, /*neg_b=*/false, b,
      /*c_mod=*/(short)0, c, /*reuse_a=*/false, /*reuse_b=*/false);
}

// A fragment (16x32 f16): lane&15 -> row; kbase = 16*(lane>>4);
// elems 0..7 = K kbase..+7, elems 8..15 = K kbase+8..+15. Two 16B loads.
static __device__ __forceinline__ v16h load_a_f16(const _Float16* __restrict__ base, int ld,
                                                  int row0, int k0, int lane) {
  const _Float16* p = base + (size_t)(row0 + (lane & 15)) * ld + k0 + ((lane >> 4) << 4);
  v16h a;
  *(v8h*)&a         = *(const v8h*)p;
  *(((v8h*)&a) + 1) = *(const v8h*)(p + 8);
  return a;
}

// B fragment (32x16) from "NT" f16 source (row n holds K contiguous):
// lane&15 -> col n; kbase = 16*(lane>>4); one contiguous 32B load.
static __device__ __forceinline__ v16h load_b_nt_f16(const _Float16* __restrict__ base, int ld,
                                                     int n0, int k0, int lane) {
  const _Float16* p = base + (size_t)(n0 + (lane & 15)) * ld + k0 + ((lane >> 4) << 4);
  return *(const v16h*)p;
}

// ---------------------------------------------------------------------------
// Kernel 0: bulk f32 -> f16 conversion (x, w_in, w_out), 8 elems/thread.
// ---------------------------------------------------------------------------
__global__ __launch_bounds__(256) void cvt_f16_kernel(const float* __restrict__ in,
                                                      _Float16* __restrict__ out, int n8) {
  int i = blockIdx.x * 256 + threadIdx.x;
  if (i >= n8) return;
  const float4* p = (const float4*)in + (size_t)i * 2;
  float4 a = p[0], b = p[1];
  v8h o;
  o[0] = (_Float16)a.x; o[1] = (_Float16)a.y; o[2] = (_Float16)a.z; o[3] = (_Float16)a.w;
  o[4] = (_Float16)b.x; o[5] = (_Float16)b.y; o[6] = (_Float16)b.z; o[7] = (_Float16)b.w;
  *((v8h*)out + i) = o;
}

// ---------------------------------------------------------------------------
// Kernel 1: fused QKV projection.  qkv[m,n] = sum_k x[m,k] * w_in[n,k] + b_in[n]
// M = B*S = 8192, N = 3E = 2304, K = E = 768.  Wave tile = 32x64, pure f16 loads.
// ---------------------------------------------------------------------------
__global__ __launch_bounds__(256) void qkv_proj_kernel(
    const _Float16* __restrict__ xh, const _Float16* __restrict__ w_in_h,
    const float* __restrict__ b_in,
    _Float16* __restrict__ q, _Float16* __restrict__ k, _Float16* __restrict__ v) {
  const int lane = threadIdx.x & 31;
  const int wave = threadIdx.x >> 5;
  const int tileN = blockIdx.x * 64;
  const int tileM = blockIdx.y * 256 + wave * 32;

  v8f acc[2][4] = {};
  for (int k0 = 0; k0 < EMB; k0 += 32) {
    v16h a0 = load_a_f16(xh, EMB, tileM, k0, lane);
    v16h a1 = load_a_f16(xh, EMB, tileM + 16, k0, lane);
#pragma unroll
    for (int nt = 0; nt < 4; ++nt) {
      v16h bf = load_b_nt_f16(w_in_h, EMB, tileN + nt * 16, k0, lane);
      acc[0][nt] = wmma_f16(a0, bf, acc[0][nt]);
      acc[1][nt] = wmma_f16(a1, bf, acc[1][nt]);
    }
  }
#pragma unroll
  for (int mt = 0; mt < 2; ++mt) {
#pragma unroll
    for (int nt = 0; nt < 4; ++nt) {
      int gn = tileN + nt * 16 + (lane & 15);
      int sec = gn / EMB;              // 0=q 1=k 2=v (16-col tile never straddles)
      int within = gn - sec * EMB;
      int h = within >> 6, d = within & 63;
      float bias = b_in[gn];
      _Float16* dst = (sec == 0) ? q : (sec == 1) ? k : v;
#pragma unroll
      for (int r = 0; r < 8; ++r) {
        int gm = tileM + mt * 16 + r + ((lane >> 4) << 3);
        int bb = gm >> 12;             // / S_LEN
        int s = gm & (S_LEN - 1);
        dst[((size_t)((bb * HEADS + h) * S_LEN + s)) * DHEAD + d] =
            (_Float16)(acc[mt][nt][r] + bias);
      }
    }
  }
}

// ---------------------------------------------------------------------------
// Kernel 2: streaming (flash) attention with anti-band mask (|i-j|<=128 masked).
// Block = 8 waves; wave owns a 16-row q block; all waves sweep k in steps of 32.
// K tile DMA'd into LDS once per block via the Tensor Data Mover (TENSORcnt);
// V tile cooperatively transposed into LDS; P round-trips LDS for C->A layout.
// ---------------------------------------------------------------------------
__global__ __launch_bounds__(256) void flash_attn_kernel(
    const _Float16* __restrict__ qg, const _Float16* __restrict__ kg,
    const _Float16* __restrict__ vg, _Float16* __restrict__ attn_out) {
  __shared__ __align__(64) _Float16 kT[32][DHEAD];   // K tile row-major     (4 KB)
  __shared__ __align__(64) _Float16 vT[DHEAD][32];   // V tile transposed    (4 KB)
  __shared__ __align__(64) _Float16 pT[8][16][32];   // per-wave P staging   (8 KB)

  const int lane = threadIdx.x & 31;
  const int wave = threadIdx.x >> 5;
  const int b = blockIdx.z, h = blockIdx.y;
  const int q0 = blockIdx.x * 128 + wave * 16;

  const size_t head_off = ((size_t)(b * HEADS + h)) * S_LEN * DHEAD;
  const _Float16* Q = qg + head_off;
  const _Float16* K = kg + head_off;
  const _Float16* V = vg + head_off;

  v16h qf[2];
  qf[0] = load_a_f16(Q, DHEAD, q0, 0, lane);
  qf[1] = load_a_f16(Q, DHEAD, q0, 32, lane);

  float mrow[8], lrow[8];
#pragma unroll
  for (int r = 0; r < 8; ++r) { mrow[r] = -1e30f; lrow[r] = 0.f; }
  v8f o[4] = {};

  const float scale = 0.125f;  // D^-0.5 = 1/8

  for (int k0 = 0; k0 < S_LEN; k0 += 32) {
    // prefetch next iteration's K/V tiles: 32 lanes x one 128B row each
    if (k0 + 32 < S_LEN) {
      __builtin_prefetch(K + (size_t)(k0 + 32 + lane) * DHEAD, 0, 3);
      __builtin_prefetch(V + (size_t)(k0 + 32 + lane) * DHEAD, 0, 3);
    }

    __syncthreads();  // previous iteration done reading kT/vT

#if USE_TDM
    if (wave == 0) {
      // D# for a 2-D tile: 32 rows x 128B of K, row stride 128B, data_size=8B.
      unsigned lds_addr = (unsigned)(size_t)&kT[0][0];
      unsigned long long ga = (unsigned long long)(size_t)(K + (size_t)k0 * DHEAD);
      v4u g0;
      g0.x = 1u;                                              // count=1 (valid user D#)
      g0.y = lds_addr;                                        // lds_addr [63:32]
      g0.z = (unsigned)(ga & 0xffffffffu);                    // global_addr [95:64]
      g0.w = (unsigned)((ga >> 32) & 0x1ffffffu) | (2u << 30);// addr [120:96] | type=2
      v8i g1;
      g1[0] = (int)(3u << 16);       // workgroup_mask=0, data_size=3 (8B)
      g1[1] = (int)(16u << 16);      // tensor_dim0 = 16 (8B units per 128B row)
      g1[2] = (int)(4096u << 16);    // tensor_dim1 = 4096 rows
      g1[3] = (int)(16u << 16);      // tile_dim0 = 16
      g1[4] = 32;                    // tile_dim1 = 32 rows (tile_dim2 = 0)
      g1[5] = 16;                    // tensor_dim0_stride = 16 (8B units)
      g1[6] = 0;
      g1[7] = 0;
      v4i gz = {0, 0, 0, 0};
#if __clang_major__ >= 23
      v8i gz8 = {0, 0, 0, 0, 0, 0, 0, 0};
      __builtin_amdgcn_tensor_load_to_lds(g0, g1, gz, gz, gz8, 0);
#else
      __builtin_amdgcn_tensor_load_to_lds(g0, g1, gz, gz, 0);
#endif
      __builtin_amdgcn_s_wait_tensorcnt(0);
    }
#else
    {  // fallback: cooperative K tile stage, 16B per thread
      int kr = threadIdx.x >> 3;
      int db = (threadIdx.x & 7) << 3;
      *(v8h*)&kT[kr][db] = *(const v8h*)(K + (size_t)(k0 + kr) * DHEAD + db);
    }
#endif

    {  // stage V[k0..k0+32, 0..64] -> vT[d][k]
      int kr = threadIdx.x >> 3;          // 0..31
      int db = (threadIdx.x & 7) << 3;    // 0..56
      const v8h vrow = *(const v8h*)(V + (size_t)(k0 + kr) * DHEAD + db);
#pragma unroll
      for (int e = 0; e < 8; ++e) vT[db + e][kr] = vrow[e];
    }
    __syncthreads();  // kT (TDM) + vT published to all waves

    // scores: S[16 x 32] = Q[16 x 64] @ K^T; B-frags from LDS (32B contiguous)
    v8f s0 = {}, s1 = {};
#pragma unroll
    for (int kk = 0; kk < 2; ++kk) {
      v16h bk0 = *(const v16h*)&kT[lane & 15][kk * 32 + ((lane >> 4) << 4)];
      v16h bk1 = *(const v16h*)&kT[16 + (lane & 15)][kk * 32 + ((lane >> 4) << 4)];
      s0 = wmma_f16(qf[kk], bk0, s0);
      s1 = wmma_f16(qf[kk], bk1, s1);
    }

    // scale + anti-band mask + per-lane tile max
    const int col0 = k0 + (lane & 15);
    const int col1 = col0 + 16;
    float tmax[8];
#pragma unroll
    for (int r = 0; r < 8; ++r) {
      int row = q0 + r + ((lane >> 4) << 3);
      float a = s0[r] * scale, c = s1[r] * scale;
      int d0 = row - col0; d0 = d0 < 0 ? -d0 : d0;
      int d1 = row - col1; d1 = d1 < 0 ? -d1 : d1;
      if (d0 <= WIN) a = -1e30f;
      if (d1 <= WIN) c = -1e30f;
      s0[r] = a; s1[r] = c;
      tmax[r] = fmaxf(a, c);
    }
    // row max across the 16 lanes of each half-wave
#pragma unroll
    for (int r = 0; r < 8; ++r) {
      float t = tmax[r];
      t = fmaxf(t, __shfl_xor(t, 1, 32));
      t = fmaxf(t, __shfl_xor(t, 2, 32));
      t = fmaxf(t, __shfl_xor(t, 4, 32));
      t = fmaxf(t, __shfl_xor(t, 8, 32));
      tmax[r] = t;
    }

    // online softmax update
    float psum[8];
#pragma unroll
    for (int r = 0; r < 8; ++r) {
      float mold = mrow[r];
      float mnew = fmaxf(mold, tmax[r]);
      bool dead = (mnew <= -5e29f);            // entire prefix masked
      float alpha = dead ? 1.0f : __expf(mold - mnew);
      float p0 = dead ? 0.0f : __expf(s0[r] - mnew);
      float p1 = dead ? 0.0f : __expf(s1[r] - mnew);
      mrow[r] = mnew;
      lrow[r] *= alpha;
      psum[r] = p0 + p1;
      s0[r] = p0; s1[r] = p1;
#pragma unroll
      for (int t = 0; t < 4; ++t) o[t][r] *= alpha;
    }
#pragma unroll
    for (int r = 0; r < 8; ++r) {
      float t = psum[r];
      t += __shfl_xor(t, 1, 32);
      t += __shfl_xor(t, 2, 32);
      t += __shfl_xor(t, 4, 32);
      t += __shfl_xor(t, 8, 32);
      lrow[r] += t;
    }

    // P: C-layout (f32) -> LDS -> A-layout (f16)
    {
      int rh = (lane >> 4) << 3;
      int c = lane & 15;
#pragma unroll
      for (int r = 0; r < 8; ++r) {
        pT[wave][r + rh][c]      = (_Float16)s0[r];
        pT[wave][r + rh][c + 16] = (_Float16)s1[r];
      }
    }
    v16h pa;
    {
      const _Float16* pp = &pT[wave][lane & 15][(lane >> 4) << 4];
      *(v8h*)&pa         = *(const v8h*)pp;
      *(((v8h*)&pa) + 1) = *(const v8h*)(pp + 8);
    }

    // O += P @ V  (4 d-tiles; B-frag = contiguous 32B read of transposed V in LDS)
#pragma unroll
    for (int t = 0; t < 4; ++t) {
      v16h vb = *(const v16h*)&vT[t * 16 + (lane & 15)][(lane >> 4) << 4];
      o[t] = wmma_f16(pa, vb, o[t]);
    }
  }

  // normalize and write [B, S, E] f16 (feeds out-proj GEMM)
#pragma unroll
  for (int r = 0; r < 8; ++r) {
    float inv = 1.0f / lrow[r];
    int row = q0 + r + ((lane >> 4) << 3);
    size_t base = ((size_t)(b * S_LEN + row)) * EMB + h * DHEAD;
#pragma unroll
    for (int t = 0; t < 4; ++t)
      attn_out[base + t * 16 + (lane & 15)] = (_Float16)(o[t][r] * inv);
  }
}

// ---------------------------------------------------------------------------
// Kernel 3: output projection. out[m,f] = sum_e attn[m,e] * w_out[f,e] + b_out[f]
// M = 8192, N = K = 768. Wave tile 32x64, pure f16 loads, f32 output.
// ---------------------------------------------------------------------------
__global__ __launch_bounds__(256) void out_proj_kernel(
    const _Float16* __restrict__ attn, const _Float16* __restrict__ w_out_h,
    const float* __restrict__ b_out, float* __restrict__ out) {
  const int lane = threadIdx.x & 31;
  const int wave = threadIdx.x >> 5;
  const int tileN = blockIdx.x * 64;
  const int tileM = blockIdx.y * 256 + wave * 32;

  v8f acc[2][4] = {};
  for (int k0 = 0; k0 < EMB; k0 += 32) {
    v16h a0 = load_a_f16(attn, EMB, tileM, k0, lane);
    v16h a1 = load_a_f16(attn, EMB, tileM + 16, k0, lane);
#pragma unroll
    for (int nt = 0; nt < 4; ++nt) {
      v16h bf = load_b_nt_f16(w_out_h, EMB, tileN + nt * 16, k0, lane);
      acc[0][nt] = wmma_f16(a0, bf, acc[0][nt]);
      acc[1][nt] = wmma_f16(a1, bf, acc[1][nt]);
    }
  }
#pragma unroll
  for (int mt = 0; mt < 2; ++mt) {
#pragma unroll
    for (int nt = 0; nt < 4; ++nt) {
      int gn = tileN + nt * 16 + (lane & 15);
      float bias = b_out[gn];
#pragma unroll
      for (int r = 0; r < 8; ++r) {
        int gm = tileM + mt * 16 + r + ((lane >> 4) << 3);
        out[(size_t)gm * EMB + gn] = acc[mt][nt][r] + bias;
      }
    }
  }
}

// ---------------------------------------------------------------------------
extern "C" void kernel_launch(void* const* d_in, const int* in_sizes, int n_in,
                              void* d_out, int out_size, void* d_ws, size_t ws_size,
                              hipStream_t stream) {
  (void)in_sizes; (void)n_in; (void)out_size; (void)ws_size;
  const float* x     = (const float*)d_in[0];
  const float* w_in  = (const float*)d_in[1];
  const float* b_in  = (const float*)d_in[2];
  const float* w_out = (const float*)d_in[3];
  const float* b_out = (const float*)d_in[4];
  float* out = (float*)d_out;

  const size_t nper   = (size_t)BATCH * HEADS * S_LEN * DHEAD;  // 6,291,456
  const size_t n_x    = (size_t)BATCH * S_LEN * EMB;            // 6,291,456
  const size_t n_win  = (size_t)3 * EMB * EMB;                  // 1,769,472
  const size_t n_wout = (size_t)EMB * EMB;                      //   589,824

  _Float16* qbuf    = (_Float16*)d_ws;
  _Float16* kbuf    = qbuf + nper;
  _Float16* vbuf    = kbuf + nper;
  _Float16* attn    = vbuf + nper;          // [B*S, E] f16
  _Float16* xh      = attn + nper;          // [B*S, E] f16
  _Float16* w_in_h  = xh + n_x;             // [3E, E] f16
  _Float16* w_out_h = w_in_h + n_win;       // [E, E] f16

  // Stage inputs to f16 once (removes all v_cvt from GEMM hot loops).
  cvt_f16_kernel<<<(int)((n_x / 8 + 255) / 256), 256, 0, stream>>>(x, xh, (int)(n_x / 8));
  cvt_f16_kernel<<<(int)((n_win / 8 + 255) / 256), 256, 0, stream>>>(w_in, w_in_h, (int)(n_win / 8));
  cvt_f16_kernel<<<(int)((n_wout / 8 + 255) / 256), 256, 0, stream>>>(w_out, w_out_h, (int)(n_wout / 8));

  // QKV: M=8192 (grid.y=32 x 256-row blocks), N=2304 (grid.x=36 x 64-col)
  qkv_proj_kernel<<<dim3(36, 32), 256, 0, stream>>>(xh, w_in_h, b_in, qbuf, kbuf, vbuf);
  // Attention: 128 q-rows per block, grid (S/128, H, B)
  flash_attn_kernel<<<dim3(S_LEN / 128, HEADS, BATCH), 256, 0, stream>>>(qbuf, kbuf, vbuf, attn);
  // Out-proj: N=768 (grid.x=12), M=8192 (grid.y=32)
  out_proj_kernel<<<dim3(12, 32), 256, 0, stream>>>(attn, w_out_h, b_out, out);
}
